// VQVAE_78597901517347
// MI455X (gfx1250) — matrix operand order repaired and checked by reading, not compiled, not run on hardware
//
#include <hip/hip_runtime.h>
#include <hip/hip_bf16.h>
#include <cstdint>

typedef __attribute__((ext_vector_type(16))) _Float16 v16h;
typedef __attribute__((ext_vector_type(8)))  float    v8f;

#define N_TOKENS 131072
#define EMB_DIM  256
#define NUM_EMB  1024
#define M_TILE   128      // tokens per workgroup (8 waves x 16 rows)
#define CHUNK    64       // codebook rows staged in LDS at a time
#define PAD_D    264      // 256 + 8 halves padding -> 528B row stride, bank-conflict free

// ---------------- block reduction helper ----------------
__device__ __forceinline__ float blockReduceSum(float v, float* sm) {
  #pragma unroll
  for (int m = 16; m >= 1; m >>= 1) v += __shfl_xor(v, m, 32);
  const int lane = threadIdx.x & 31;
  const int wv   = threadIdx.x >> 5;
  if (lane == 0) sm[wv] = v;
  __syncthreads();
  const int nw = blockDim.x >> 5;
  v = (threadIdx.x < (unsigned)nw) ? sm[threadIdx.x] : 0.0f;
  if (wv == 0) {
    #pragma unroll
    for (int m = 16; m >= 1; m >>= 1) v += __shfl_xor(v, m, 32);
  }
  return v;  // valid on thread 0
}

// ---------------- 1) codebook f32->f16 + norms, zero scratch ----------------
__global__ __launch_bounds__(256) void vq_prep(const float* __restrict__ cb,
                                               _Float16* __restrict__ cb16,
                                               float* __restrict__ cbn,
                                               int* __restrict__ hist,
                                               float* __restrict__ sse) {
  __shared__ float sm[32];
  const int row = blockIdx.x;
  const int tid = threadIdx.x;
  const float v = cb[(size_t)row * EMB_DIM + tid];
  cb16[(size_t)row * EMB_DIM + tid] = (_Float16)v;
  const float s = blockReduceSum(v * v, sm);
  if (tid == 0) {
    cbn[row]  = s;
    hist[row] = 0;
    if (row == 0) sse[0] = 0.0f;
  }
}

// ---------------- 2) WMMA distance + argmin ----------------
// dist[m,k] = ||e_k||^2 - 2 * x_m . e_k   (||x_m||^2 dropped: constant per row)
__global__ __launch_bounds__(256) void vq_argmin(const float* __restrict__ x,
                                                 const _Float16* __restrict__ cb16,
                                                 const float* __restrict__ cbn,
                                                 int* __restrict__ outidx) {
  __shared__ _Float16 lcb[CHUNK * PAD_D];
  __shared__ float    lnorm[CHUNK];

  const int tid  = threadIdx.x;
  const int lane = tid & 31;
  const int wv   = tid >> 5;       // 8 waves
  const int hl   = lane >> 4;      // half-wave: 0 or 1
  const int l15  = lane & 15;
  const int rowBase = blockIdx.x * M_TILE + wv * 16;
  const int m = rowBase + l15;     // this lane's A row

  // Load A strip (16 rows x 256 dims) into v16h fragments, f32 -> f16.
  // A 16x32 in this code's contiguous k order (same as B): elems 0..15 = K(off..off+15),
  // off = 0 for lanes 0-15, 16 for lanes 16-31.
  v16h a[8];
  const float* xr = x + (size_t)m * EMB_DIM;
  const int off = hl ? 16 : 0;
  #pragma unroll
  for (int s = 0; s < 8; ++s) {
    const int d0 = 32 * s + off;
    #pragma unroll
    for (int j = 0; j < 8; ++j) {
      a[s][j]     = (_Float16)xr[d0 + j];
      a[s][8 + j] = (_Float16)xr[d0 + 8 + j];
    }
  }

  float minv[8];
  int   mini[8];
  #pragma unroll
  for (int r = 0; r < 8; ++r) { minv[r] = 3.4e38f; mini[r] = 0; }

  // LDS byte offset of the staging buffer (flat LDS aperture: addr[31:0] == LDS offset)
  const uint32_t lcb_base = (uint32_t)(size_t)(&lcb[0]);

  for (int ch = 0; ch < NUM_EMB / CHUNK; ++ch) {
    __syncthreads();   // previous chunk fully consumed before overwrite
    // --- async stage: CHUNK rows x 256 halves (32 KB) -> padded LDS ---
    // global_load_async_to_lds_b128 (GVS): per lane 16B memory -> LDS, ASYNCcnt.
    {
      #pragma unroll
      for (int it = 0; it < (CHUNK * EMB_DIM * 2) / (256 * 16); ++it) {  // 8 iters
        const int i = tid + it * 256;            // uint4 index within chunk
        const int r = i >> 5;                    // 32 uint4 per 256-half row
        const int c = i & 31;
        const uint32_t voff  = (uint32_t)ch * (CHUNK * EMB_DIM * 2) + (uint32_t)i * 16u;
        const uint32_t laddr = lcb_base + (uint32_t)(r * (PAD_D * 2) + c * 16);
        asm volatile("global_load_async_to_lds_b128 %0, %1, %2"
                     :: "v"(laddr), "v"(voff), "s"(cb16) : "memory");
      }
      for (int i = tid; i < CHUNK; i += 256) lnorm[i] = cbn[ch * CHUNK + i];
      asm volatile("s_wait_asynccnt 0x0" ::: "memory");
    }
    __syncthreads();   // LDS writes visible to all 8 waves

    #pragma unroll
    for (int t = 0; t < CHUNK / 16; ++t) {
      const int cl = t * 16 + l15;                    // local code (= B column)
      // B 32x16 in the same contiguous k order as A: lane holds 16 contiguous K-values of its column,
      // lanes 0-15 -> K 0..15, lanes 16-31 -> K 16..31 (per k-step).
      const _Float16* bp = lcb + cl * PAD_D + (hl ? 16 : 0);
      // preload all 8 B fragments, then issue WMMAs as two independent chains
      v16h b[8];
      #pragma unroll
      for (int s = 0; s < 8; ++s) {
        union { uint4 q[2]; v16h v; } bu;
        const uint4* p = (const uint4*)(bp + 32 * s); // 16B-aligned LDS read
        bu.q[0] = p[0];
        bu.q[1] = p[1];
        b[s] = bu.v;
      }
      v8f acc0 = {0.f, 0.f, 0.f, 0.f, 0.f, 0.f, 0.f, 0.f};
      v8f acc1 = {0.f, 0.f, 0.f, 0.f, 0.f, 0.f, 0.f, 0.f};
      #pragma unroll
      for (int s = 0; s < 8; s += 2) {
        acc0 = __builtin_amdgcn_wmma_f32_16x16x32_f16(
            false, a[s],     false, b[s],     (short)0, acc0, false, false);
        acc1 = __builtin_amdgcn_wmma_f32_16x16x32_f16(
            false, a[s + 1], false, b[s + 1], (short)0, acc1, false, false);
      }
      const int   cglob = ch * CHUNK + cl;
      const float nrm   = lnorm[cl];
      // C layout: slot r = row (rowBase + r + 8*hl), col = tile base + l15
      #pragma unroll
      for (int r = 0; r < 8; ++r) {
        const float d = nrm - 2.0f * (acc0[r] + acc1[r]);
        if (d < minv[r]) { minv[r] = d; mini[r] = cglob; }
      }
    }
  }

  // argmin across the 16 lanes of each half-wave (columns), first-index ties
  #pragma unroll
  for (int r = 0; r < 8; ++r) {
    float v = minv[r];
    int   ix = mini[r];
    #pragma unroll
    for (int msk = 8; msk >= 1; msk >>= 1) {
      const float ov = __shfl_xor(v, msk, 16);
      const int   oi = __shfl_xor(ix, msk, 16);
      if (ov < v || (ov == v && oi < ix)) { v = ov; ix = oi; }
    }
    if (l15 == 0) outidx[rowBase + r + hl * 8] = ix;
  }
}

// ---------------- 3) gather quantized rows + SSE ----------------
__global__ __launch_bounds__(256) void vq_gather(const float* __restrict__ x,
                                                 const float* __restrict__ cb,
                                                 const int* __restrict__ idx,
                                                 float* __restrict__ outq,
                                                 float* __restrict__ sse,
                                                 int total) {
  __shared__ float sm[32];
  float acc = 0.0f;
  for (int i = blockIdx.x * blockDim.x + threadIdx.x; i < total;
       i += gridDim.x * blockDim.x) {
    const int tok = i >> 8;
    const int d   = i & 255;
    const float q = cb[(size_t)idx[tok] * EMB_DIM + d];
    outq[i] = q;  // quantized_st == quantized numerically
    const float diff = q - x[i];
    acc += diff * diff;
  }
  const float s = blockReduceSum(acc, sm);
  if (threadIdx.x == 0) atomicAdd(sse, s);
}

// ---------------- 4) LDS-privatized histogram ----------------
__global__ __launch_bounds__(256) void vq_hist(const int* __restrict__ idx,
                                               int* __restrict__ hist, int n) {
  __shared__ int lh[NUM_EMB];
  for (int i = threadIdx.x; i < NUM_EMB; i += blockDim.x) lh[i] = 0;
  __syncthreads();
  for (int i = blockIdx.x * blockDim.x + threadIdx.x; i < n;
       i += gridDim.x * blockDim.x)
    atomicAdd(&lh[idx[i]], 1);
  __syncthreads();
  for (int i = threadIdx.x; i < NUM_EMB; i += blockDim.x)
    if (lh[i]) atomicAdd(&hist[i], lh[i]);
}

// ---------------- 5) finalize: loss + perplexity ----------------
__global__ __launch_bounds__(1024) void vq_final(const int* __restrict__ hist,
                                                 const float* __restrict__ sse,
                                                 float* __restrict__ out,
                                                 int nd_total) {
  __shared__ float sm[32];
  const int tid = threadIdx.x;
  const float p = (float)hist[tid] * (1.0f / (float)N_TOKENS);
  const float e = -p * logf(p + 1e-10f);
  const float s = blockReduceSum(e, sm);
  if (tid == 0) {
    // loss = q_latent + 0.25*e_latent, both == mean((q-x)^2)
    out[0] = 1.25f * sse[0] / (float)nd_total;
    out[1 + nd_total] = expf(s);
  }
}

extern "C" void kernel_launch(void* const* d_in, const int* in_sizes, int n_in,
                              void* d_out, int out_size, void* d_ws, size_t ws_size,
                              hipStream_t stream) {
  const float* x  = (const float*)d_in[0];  // [131072, 256]
  const float* cb = (const float*)d_in[1];  // [1024, 256]
  float* out = (float*)d_out;               // [loss, quantized(33554432), perplexity]

  // workspace carve-out (~1.06 MB)
  char* w = (char*)d_ws;
  _Float16* cb16 = (_Float16*)(w);                           // 524288 B
  float*    cbn  = (float*)(w + 524288);                     //   4096 B
  int*      ib   = (int*)(w + 524288 + 4096);                // 524288 B
  int*      hist = (int*)(w + 1052672);                      //   4096 B
  float*    sse  = (float*)(w + 1056768);                    //     16 B

  const int total = N_TOKENS * EMB_DIM;

  vq_prep  <<<NUM_EMB,            256, 0, stream>>>(cb, cb16, cbn, hist, sse);
  vq_argmin<<<N_TOKENS / M_TILE,  256, 0, stream>>>(x, cb16, cbn, ib);
  vq_gather<<<4096,               256, 0, stream>>>(x, cb, ib, out + 1, sse, total);
  vq_hist  <<<512,                256, 0, stream>>>(ib, hist, N_TOKENS);
  vq_final <<<1,                 1024, 0, stream>>>(hist, sse, out, total);
}
